// RGSLCell_30279519437073
// MI455X (gfx1250) — compile-verified, matrix operand order checked by reading, not run on hardware
//
#include <hip/hip_runtime.h>
#include <hip/hip_bf16.h>

typedef __bf16 bf16;
typedef __attribute__((ext_vector_type(16))) __bf16 v16bf;
typedef __attribute__((ext_vector_type(8)))  float  v8f;

#define LEAK 0.01f

static const int Bb = 64, Nn = 1024, DIN = 32, DOUT = 64, Kk = 3, ED = 16, CIN = 96;
static const long BN = (long)Bb * Nn;   // 65536 rows

// ---- CDNA5 async global->LDS staging (guarded probe) -----------------------
#if defined(__has_builtin)
#  if __has_builtin(__builtin_amdgcn_global_load_async_to_lds_b128)
#    define HAVE_ASYNC_LDS 1
#  endif
#endif
#ifndef HAVE_ASYNC_LDS
#  define HAVE_ASYNC_LDS 0
#endif

typedef __attribute__((__vector_size__(4 * sizeof(int)))) int vint4;  // native vector
typedef __attribute__((address_space(1))) vint4 gas_v4i;   // global
typedef __attribute__((address_space(3))) vint4 las_v4i;   // LDS

__device__ __forceinline__ void wait_asynccnt0() {
#if defined(__has_builtin) && __has_builtin(__builtin_amdgcn_s_wait_asynccnt)
  __builtin_amdgcn_s_wait_asynccnt(0);
#else
  asm volatile("s_wait_asynccnt 0" ::: "memory");
#endif
}

// ---------------------------------------------------------------------------
// Generic tiled bf16 GEMM with f32 WMMA accumulation.
//   C[m, n] = sum_k A[m, k] * B[k, n]   (+ epilogue)
// epi: 0 = store bf16
//      1 = store bf16 of (2*acc - I)          (chebyshev s2)
//      2 = store bf16 of (acc + bias[col])
//      3 = store f32  of leaky(acc + bias[col])
// ---------------------------------------------------------------------------
#define BM_T 128
#define BN_T 128
#define BK_T 32
#define PADK 40        // BK_T + 8 bf16 elements of padding per LDS row

union Frag { uint4 q[2]; v16bf v; };

__device__ __forceinline__ v8f wmma_bf16(v16bf a, v16bf b, v8f c) {
  return __builtin_amdgcn_wmma_f32_16x16x32_bf16(false, a, false, b, (short)0, c,
                                                 false, false);
}

__global__ void __launch_bounds__(256)
k_gemm_bf16(const bf16* __restrict__ A, long lda, long strideA,
            const bf16* __restrict__ Bm, long ldb, long strideB,
            void* __restrict__ Cv, long ldc, long strideC,
            const float* __restrict__ bias, long strideBias,
            int M, int Nc, int Kc, int epi)
{
  __shared__ bf16 sA[2][BM_T * PADK];
  __shared__ bf16 sB[2][BN_T * PADK];

  const int tid   = threadIdx.x;
  const int lane  = tid & 31;
  const int wave  = tid >> 5;
  const int waveM = wave >> 1;          // 0..3 -> 32-row strip
  const int waveN = wave & 1;           // 0..1 -> 64-col strip
  const int z     = blockIdx.z;
  const int tileM = blockIdx.x * BM_T;
  const int tileN = blockIdx.y * BN_T;

  A  += (long)z * strideA;
  Bm += (long)z * strideB;
  bf16*  Cb16 = (bf16*)Cv;
  float* Cf32 = (float*)Cv;
  if (epi >= 3) Cf32 += (long)z * strideC;
  else          Cb16 += (long)z * strideC;
  if (bias) bias += (long)z * strideBias;

  v8f acc[2][4];
  for (int i = 0; i < 2; ++i)
    for (int j = 0; j < 4; ++j)
      for (int r = 0; r < 8; ++r) acc[i][j][r] = 0.f;

  const int nK = Kc / BK_T;             // all Kc are multiples of 32

  // ---- B tile staging: global -> regs -> transposed LDS [n][k] ------------
  uint4 regB[2];
  int bK[2], bN[2];
  auto loadB = [&](int kt) {
    const int k0 = kt * BK_T;
    #pragma unroll
    for (int it = 0; it < 2; ++it) {
      int idx = tid + it * 256;                 // 0..511
      bK[it] = idx >> 4; bN[it] = (idx & 15) << 3;
      int gn = tileN + bN[it];
      const bf16* bs = Bm + (long)(k0 + bK[it]) * ldb + gn;
      if (gn + 7 < Nc) {
        regB[it] = *(const uint4*)bs;
      } else {
        union { uint4 q; bf16 h[8]; } u;
        u.q.x = u.q.y = u.q.z = u.q.w = 0u;
        for (int j = 0; j < 8; ++j) if (gn + j < Nc) u.h[j] = bs[j];
        regB[it] = u.q;
      }
      if (it == 0 && kt + 2 < nK)
        __builtin_prefetch(bs + (long)(2 * BK_T) * ldb, 0, 1);
    }
  };
  auto storeB = [&](int buf) {
    #pragma unroll
    for (int it = 0; it < 2; ++it) {
      union { uint4 q; bf16 h[8]; } u; u.q = regB[it];
      #pragma unroll
      for (int j = 0; j < 8; ++j)
        sB[buf][(bN[it] + j) * PADK + bK[it]] = u.h[j];   // transpose -> [n][k]
    }
  };

  // ---- A tile staging ------------------------------------------------------
#if HAVE_ASYNC_LDS
  // async DMA straight into LDS; OOB rows clamp to a valid row (their D rows
  // are never stored: epilogue masks m < M; WMMA row m depends only on A row m)
  auto stageA = [&](int kt, int buf) {
    const int k0 = kt * BK_T;
    #pragma unroll
    for (int it = 0; it < 2; ++it) {
      int idx = tid + it * 256;
      int row = idx >> 2, c8 = (idx & 3) << 3;
      int gr = tileM + row; if (gr >= M) gr = tileM < M ? tileM : 0;
      const bf16* src = A + (long)gr * lda + (k0 + c8);
      bf16* dst = &sA[buf][row * PADK + c8];
      __builtin_amdgcn_global_load_async_to_lds_b128(
          (gas_v4i*)(void*)src,
          (las_v4i*)(void*)dst,
          0, 0);
    }
  };
#else
  uint4 regA[2];
  int aRow[2], aOff[2];
  auto loadA = [&](int kt) {
    const int k0 = kt * BK_T;
    #pragma unroll
    for (int it = 0; it < 2; ++it) {
      int idx = tid + it * 256;
      aRow[it] = idx >> 2; aOff[it] = (idx & 3) << 3;
      int gr = tileM + aRow[it];
      if (gr < M) {
        regA[it] = *(const uint4*)(A + (long)gr * lda + (k0 + aOff[it]));
      } else {
        regA[it].x = regA[it].y = regA[it].z = regA[it].w = 0u;
      }
    }
  };
  auto storeA = [&](int buf) {
    #pragma unroll
    for (int it = 0; it < 2; ++it)
      *(uint4*)&sA[buf][aRow[it] * PADK + aOff[it]] = regA[it];
  };
#endif

  // ---- prologue ------------------------------------------------------------
#if HAVE_ASYNC_LDS
  loadB(0);
  stageA(0, 0);
  storeB(0);
  wait_asynccnt0();
#else
  loadA(0);
  loadB(0);
  storeA(0);
  storeB(0);
#endif
  __syncthreads();

  const int half8 = (lane >> 4) << 3;   // K striping: lanes 16..31 start at K=8
  const int l16   = lane & 15;

  for (int kt = 0; kt < nK; ++kt) {
    const int buf = kt & 1;
#if HAVE_ASYNC_LDS
    if (kt + 1 < nK) { loadB(kt + 1); stageA(kt + 1, buf ^ 1); }
#else
    if (kt + 1 < nK) { loadA(kt + 1); loadB(kt + 1); }
#endif

    Frag fa[2], fb[4];
    #pragma unroll
    for (int tm = 0; tm < 2; ++tm) {
      const bf16* p = &sA[buf][(waveM * 32 + tm * 16 + l16) * PADK + half8];
      fa[tm].q[0] = *(const uint4*)p;        // K = half8..half8+7
      fa[tm].q[1] = *(const uint4*)(p + 16); // K = 16+half8..
    }
    #pragma unroll
    for (int tn = 0; tn < 4; ++tn) {
      const bf16* p = &sB[buf][(waveN * 64 + tn * 16 + l16) * PADK + half8];
      fb[tn].q[0] = *(const uint4*)p;
      fb[tn].q[1] = *(const uint4*)(p + 16);
    }
    #pragma unroll
    for (int tm = 0; tm < 2; ++tm)
      #pragma unroll
      for (int tn = 0; tn < 4; ++tn)
        acc[tm][tn] = wmma_bf16(fa[tm].v, fb[tn].v, acc[tm][tn]);

#if HAVE_ASYNC_LDS
    wait_asynccnt0();                    // our async writes into buf^1 done
    __syncthreads();
    if (kt + 1 < nK) storeB(buf ^ 1);
#else
    __syncthreads();
    if (kt + 1 < nK) { storeA(buf ^ 1); storeB(buf ^ 1); }
#endif
    __syncthreads();
  }

  // epilogue: C/D layout -> element r: m_local = r + (lane>=16 ? 8 : 0), n_local = lane&15
  #pragma unroll
  for (int tm = 0; tm < 2; ++tm) {
    #pragma unroll
    for (int tn = 0; tn < 4; ++tn) {
      int nCol = tileN + waveN * 64 + tn * 16 + l16;
      if (nCol >= Nc) continue;
      float bv = bias ? bias[nCol] : 0.f;
      int mBase = tileM + waveM * 32 + tm * 16 + half8;
      #pragma unroll
      for (int r = 0; r < 8; ++r) {
        int m = mBase + r;
        if (m >= M) continue;
        float v = acc[tm][tn][r];
        long off = (long)m * ldc + nCol;
        switch (epi) {
          case 0: Cb16[off] = (bf16)v; break;
          case 1: Cb16[off] = (bf16)(2.f * v - (m == nCol ? 1.f : 0.f)); break;
          case 2: Cb16[off] = (bf16)(v + bv); break;
          default: { float t = v + bv; Cf32[off] = t >= 0.f ? t : LEAK * t; } break;
        }
      }
    }
  }
}

// ------------------------------- small kernels ------------------------------
__global__ void k_f2bf(const float* __restrict__ s, bf16* __restrict__ d, long n) {
  long i = (long)blockIdx.x * blockDim.x + threadIdx.x;
  if (i < n) d[i] = (bf16)s[i];
}

// out[n, j] = sum_d emb[n, d] * pool[d, j]   (bf16 out)
__global__ void k_mixw(const float* __restrict__ emb, const float* __restrict__ pool,
                       bf16* __restrict__ out, int J) {
  int n = blockIdx.y;
  int j = blockIdx.x * blockDim.x + threadIdx.x;
  if (j >= J) return;
  float s = 0.f;
  #pragma unroll
  for (int d = 0; d < ED; ++d) s += emb[n * ED + d] * pool[(long)d * J + j];
  out[(long)n * J + j] = (bf16)s;
}

// f32 version (bias pools)
__global__ void k_mixb(const float* __restrict__ emb, const float* __restrict__ pool,
                       float* __restrict__ out, int J) {
  int n = blockIdx.y;
  int j = blockIdx.x * blockDim.x + threadIdx.x;
  if (j >= J) return;
  float s = 0.f;
  #pragma unroll
  for (int d = 0; d < ED; ++d) s += emb[n * ED + d] * pool[(long)d * J + j];
  out[(long)n * J + j] = s;
}

// wt[i, o] = (bf16) w[o, i]   (w is [O, I] row-major)
__global__ void k_transpose(const float* __restrict__ w, bf16* __restrict__ wt,
                            int O, int I) {
  int idx = blockIdx.x * blockDim.x + threadIdx.x;
  if (idx >= O * I) return;
  int o = idx / I, i = idx % I;
  wt[(long)i * O + o] = (bf16)w[idx];
}

// Xg = bf16(concat(x, state)); also write the k=0 slice of the x_g buffer
__global__ void k_build_xg(const float* __restrict__ x, const float* __restrict__ st,
                           bf16* __restrict__ Xg, bf16* __restrict__ XGk0) {
  long row = blockIdx.x;
  int c = threadIdx.x;                    // 0..95
  float v = (c < DIN) ? x[row * DIN + c] : st[row * DOUT + (c - DIN)];
  bf16 h = (bf16)v;
  Xg[row * CIN + c] = h;
  XGk0[row * (Kk * CIN) + c] = h;
}

// column mean over n (AdaptiveAvgPool1d(1) over the node axis)
__global__ void k_mean(const float* __restrict__ src, float* __restrict__ dst, int O) {
  int b = blockIdx.x;
  int o = threadIdx.x;
  float s = 0.f;
  for (int n = 0; n < Nn; ++n) s += src[((long)b * Nn + n) * O + o];
  dst[b * O + o] = s * (1.f / (float)Nn);
}

// per-batch scalar gates: g[b] = sigmoid(relu(mean @ w1^T) @ w2^T)
__global__ void k_att_gate(const float* __restrict__ m0, const float* __restrict__ m1,
                           const float* __restrict__ w1a, const float* __restrict__ w2a,
                           const float* __restrict__ w1b, const float* __restrict__ w2b,
                           float* __restrict__ g, int O, int H) {
  int b = blockIdx.x * blockDim.x + threadIdx.x;
  if (b >= Bb) return;
  float s = 0.f;
  for (int j = 0; j < H; ++j) {
    float a = 0.f;
    for (int o = 0; o < O; ++o) a += m0[b * O + o] * w1a[j * O + o];
    s += fmaxf(a, 0.f) * w2a[j];
  }
  g[b] = 1.f / (1.f + expf(-s));
  s = 0.f;
  for (int j = 0; j < H; ++j) {
    float a = 0.f;
    for (int o = 0; o < O; ++o) a += m1[b * O + o] * w1b[j * O + o];
    s += fmaxf(a, 0.f) * w2b[j];
  }
  g[Bb + b] = 1.f / (1.f + expf(-s));
}

// z_r = sigmoid(l0*g1[b] + l1*g2[b]); build candidate input Xu=concat(x, z*state);
// stash r for the final combine; also refresh the k=0 slice of the x_g buffer.
__global__ void k_gate_combine(const float* __restrict__ L0, const float* __restrict__ L1,
                               const float* __restrict__ g, const float* __restrict__ x,
                               const float* __restrict__ st,
                               bf16* __restrict__ Xu, bf16* __restrict__ XGk0,
                               float* __restrict__ Rb) {
  long row = blockIdx.x;
  int o = threadIdx.x;                    // 0..127
  int b = (int)(row >> 10);
  float zr = L0[row * 128 + o] * g[b] + L1[row * 128 + o] * g[Bb + b];
  zr = 1.f / (1.f + expf(-zr));
  if (o < DOUT) {
    float v = zr * st[row * DOUT + o];
    bf16 h = (bf16)v;
    Xu[row * CIN + DIN + o] = h;
    XGk0[row * (Kk * CIN) + DIN + o] = h;
    if (o < DIN) {
      bf16 hx = (bf16)x[row * DIN + o];
      Xu[row * CIN + o] = hx;
      XGk0[row * (Kk * CIN) + o] = hx;
    }
  } else {
    Rb[row * DOUT + (o - DOUT)] = zr;
  }
}

// h = r*state + (1-r)*tanh(l0*g1[b] + l1*g2[b])
__global__ void k_final(const float* __restrict__ L0, const float* __restrict__ L1,
                        const float* __restrict__ g, const float* __restrict__ r,
                        const float* __restrict__ st, float* __restrict__ out) {
  long row = blockIdx.x;
  int o = threadIdx.x;                    // 0..63
  int b = (int)(row >> 10);
  float hc = tanhf(L0[row * DOUT + o] * g[b] + L1[row * DOUT + o] * g[Bb + b]);
  float rv = r[row * DOUT + o];
  out[row * DOUT + o] = rv * st[row * DOUT + o] + (1.f - rv) * hc;
}

// --------------------------------- host -------------------------------------
extern "C" void kernel_launch(void* const* d_in, const int* in_sizes, int n_in,
                              void* d_out, int out_size, void* d_ws, size_t ws_size,
                              hipStream_t stream) {
  (void)in_sizes; (void)n_in; (void)out_size; (void)ws_size;
  const float* x      = (const float*)d_in[0];
  const float* st     = (const float*)d_in[1];
  const float* emb    = (const float*)d_in[2];
  const float* Lt     = (const float*)d_in[3];
  const float* cheb   = (const float*)d_in[4];
  const float* g_wpool= (const float*)d_in[5];
  const float* g_bpool= (const float*)d_in[6];
  const float* g_iw   = (const float*)d_in[7];
  const float* g_ib   = (const float*)d_in[8];
  const float* g_gw   = (const float*)d_in[9];
  const float* g_gb   = (const float*)d_in[10];
  const float* g_a1w1 = (const float*)d_in[11];
  const float* g_a1w2 = (const float*)d_in[12];
  const float* g_a2w1 = (const float*)d_in[13];
  const float* g_a2w2 = (const float*)d_in[14];
  const float* u_wpool= (const float*)d_in[15];
  const float* u_bpool= (const float*)d_in[16];
  const float* u_iw   = (const float*)d_in[17];
  const float* u_ib   = (const float*)d_in[18];
  const float* u_gw   = (const float*)d_in[19];
  const float* u_gb   = (const float*)d_in[20];
  const float* u_a1w1 = (const float*)d_in[21];
  const float* u_a1w2 = (const float*)d_in[22];
  const float* u_a2w1 = (const float*)d_in[23];
  const float* u_a2w2 = (const float*)d_in[24];
  float* out = (float*)d_out;

  const long NN = 1024L * 1024L;
  char* ws = (char*)d_ws;
  size_t off = 0;
  auto alloc = [&](size_t bytes) {
    off = (off + 255) & ~(size_t)255;
    char* p = ws + off; off += bytes; return p;
  };
  bf16*  S1    = (bf16*)alloc(NN * 2);                 // bf16(L)
  bf16*  S2    = (bf16*)alloc(NN * 2);                 // bf16(2*L@L - I)
  bf16*  C16   = (bf16*)alloc(3 * NN * 2);             // bf16(cheb)
  bf16*  W     = (bf16*)alloc((size_t)Nn * 36864 * 2); // dyn weights (gate then upd)
  bf16*  Xg    = (bf16*)alloc((size_t)BN * 96 * 2);    // concat input (gate then upd)
  bf16*  XG    = (bf16*)alloc((size_t)BN * 288 * 2);   // x_g [b,n,3*96]
  bf16*  X0    = (bf16*)alloc((size_t)BN * 128 * 2);   // init_gconv out
  bf16*  XG1   = (bf16*)alloc((size_t)BN * 384 * 2);   // x_g1 [b,n,3*O]
  float* L0    = (float*)alloc((size_t)BN * 128 * 4);  // leaky(gconv0)
  float* L1    = (float*)alloc((size_t)BN * 128 * 4);  // leaky(gconv1)
  float* Rb    = (float*)alloc((size_t)BN * 64 * 4);   // r gate
  float* biasg = (float*)alloc((size_t)Nn * 128 * 4);
  float* biasu = (float*)alloc((size_t)Nn * 64 * 4);
  bf16*  iwTg  = (bf16*)alloc(96 * 128 * 2);
  bf16*  gwTg  = (bf16*)alloc(384 * 128 * 2);
  bf16*  iwTu  = (bf16*)alloc(96 * 64 * 2);
  bf16*  gwTu  = (bf16*)alloc(192 * 64 * 2);
  float* mean0 = (float*)alloc(64 * 128 * 4);
  float* mean1 = (float*)alloc(64 * 128 * 4);
  float* gateG = (float*)alloc(128 * 4);
  float* gateU = (float*)alloc(128 * 4);

  auto gemm = [&](const bf16* A, long lda, long sA, const bf16* Bp, long ldb, long sB,
                  void* C, long ldc, long sC, const float* bias, long sBias,
                  int M, int Nc, int Kc, int epi, int gz) {
    dim3 grid((M + BM_T - 1) / BM_T, (Nc + BN_T - 1) / BN_T, gz);
    k_gemm_bf16<<<grid, 256, 0, stream>>>(A, lda, sA, Bp, ldb, sB,
                                          C, ldc, sC, bias, sBias, M, Nc, Kc, epi);
  };

  // ---- supports -----------------------------------------------------------
  k_f2bf<<<(unsigned)((NN + 255) / 256), 256, 0, stream>>>(Lt, S1, NN);
  k_f2bf<<<(unsigned)((3 * NN + 255) / 256), 256, 0, stream>>>(cheb, C16, 3 * NN);
  gemm(S1, 1024, 0, S1, 1024, 0, S2, 1024, 0, nullptr, 0, 1024, 1024, 1024, 1, 1);

  // ---- gate path ----------------------------------------------------------
  k_build_xg<<<(unsigned)BN, 96, 0, stream>>>(x, st, Xg, XG);
  k_mixw<<<dim3(36864 / 256, Nn), 256, 0, stream>>>(emb, g_wpool, W, 36864);
  k_mixb<<<dim3(1, Nn), 256, 0, stream>>>(emb, g_bpool, biasg, 128);
  k_transpose<<<(128 * 96 + 255) / 256, 256, 0, stream>>>(g_iw, iwTg, 128, 96);
  k_transpose<<<(128 * 384 + 255) / 256, 256, 0, stream>>>(g_gw, gwTg, 128, 384);

  // x_g slices k=1,2 : [N,N] @ [N,96] per batch
  gemm(S1, 1024, 0, Xg, 96, (long)Nn * 96, XG + 96,  288, (long)Nn * 288,
       nullptr, 0, 1024, 96, 1024, 0, Bb);
  gemm(S2, 1024, 0, Xg, 96, (long)Nn * 96, XG + 192, 288, (long)Nn * 288,
       nullptr, 0, 1024, 96, 1024, 0, Bb);
  // per-node dynamic gconv0 -> leaky -> L0 (grid.z = node)
  gemm(XG, (long)Nn * 288, 288, W, 128, 36864, L0, (long)Nn * 128, 128,
       biasg, 128, Bb, 128, 288, 3, Nn);
  // x0 = Xg @ iw^T + ib
  gemm(Xg, 96, 0, iwTg, 128, 0, X0, 128, 0, g_ib, 0, (int)BN, 128, 96, 2, 1);
  // x_g1 slices k=0..2 : cheb_k @ x0 per batch
  for (int k = 0; k < 3; ++k)
    gemm(C16 + (long)k * NN, 1024, 0, X0, 128, (long)Nn * 128,
         XG1 + k * 128, 384, (long)Nn * 384, nullptr, 0, 1024, 128, 1024, 0, Bb);
  // gconv1 -> leaky -> L1
  gemm(XG1, 384, 0, gwTg, 128, 0, L1, 128, 0, g_gb, 0, (int)BN, 128, 384, 3, 1);

  k_mean<<<Bb, 128, 0, stream>>>(L0, mean0, 128);
  k_mean<<<Bb, 128, 0, stream>>>(L1, mean1, 128);
  k_att_gate<<<2, 32, 0, stream>>>(mean0, mean1, g_a1w1, g_a1w2, g_a2w1, g_a2w2,
                                   gateG, 128, 8);
  k_gate_combine<<<(unsigned)BN, 128, 0, stream>>>(L0, L1, gateG, x, st, Xg, XG, Rb);

  // ---- update path --------------------------------------------------------
  k_mixw<<<dim3(18432 / 256, Nn), 256, 0, stream>>>(emb, u_wpool, W, 18432);
  k_mixb<<<dim3(1, Nn), 256, 0, stream>>>(emb, u_bpool, biasu, 64);
  k_transpose<<<(64 * 96 + 255) / 256, 256, 0, stream>>>(u_iw, iwTu, 64, 96);
  k_transpose<<<(64 * 192 + 255) / 256, 256, 0, stream>>>(u_gw, gwTu, 64, 192);

  gemm(S1, 1024, 0, Xg, 96, (long)Nn * 96, XG + 96,  288, (long)Nn * 288,
       nullptr, 0, 1024, 96, 1024, 0, Bb);
  gemm(S2, 1024, 0, Xg, 96, (long)Nn * 96, XG + 192, 288, (long)Nn * 288,
       nullptr, 0, 1024, 96, 1024, 0, Bb);
  gemm(XG, (long)Nn * 288, 288, W, 64, 18432, L0, (long)Nn * 64, 64,
       biasu, 64, Bb, 64, 288, 3, Nn);
  gemm(Xg, 96, 0, iwTu, 64, 0, X0, 64, 0, u_ib, 0, (int)BN, 64, 96, 2, 1);
  for (int k = 0; k < 3; ++k)
    gemm(C16 + (long)k * NN, 1024, 0, X0, 64, (long)Nn * 64,
         XG1 + k * 64, 192, (long)Nn * 192, nullptr, 0, 1024, 64, 1024, 0, Bb);
  gemm(XG1, 192, 0, gwTu, 64, 0, L1, 64, 0, u_gb, 0, (int)BN, 64, 192, 3, 1);

  k_mean<<<Bb, 64, 0, stream>>>(L0, mean0, 64);
  k_mean<<<Bb, 64, 0, stream>>>(L1, mean1, 64);
  k_att_gate<<<2, 32, 0, stream>>>(mean0, mean1, u_a1w1, u_a1w2, u_a2w1, u_a2w2,
                                   gateU, 64, 4);
  k_final<<<(unsigned)BN, 64, 0, stream>>>(L0, L1, gateU, Rb, st, out);
}
